// _CoreAttention_25812753449408
// MI455X (gfx1250) — 16-bit and 8-bit matrix operand order repaired and checked by reading, compiled, not run on hardware
//
#include <hip/hip_runtime.h>
#include <hip/hip_bf16.h>
#include <math.h>

typedef __attribute__((ext_vector_type(8)))  int      v8i;
typedef __attribute__((ext_vector_type(8)))  float    v8f;
typedef __attribute__((ext_vector_type(16))) _Float16 v16h;
typedef __attribute__((ext_vector_type(4)))  _Float16 h4;

#define S_LEN   2048
#define BH      32
#define DIM     128
#define QT      64
#define KT      64
#define NW      4
#define SCALE_LOG2 0.12753420904971635f

__device__ __forceinline__ unsigned pack4(int4 a) {
  return (unsigned)(a.x & 255) | ((unsigned)(a.y & 255) << 8) |
         ((unsigned)(a.z & 255) << 16) | ((unsigned)(a.w & 255) << 24);
}

__global__ __launch_bounds__(128)
void flash_attn_u8_kernel(const int* __restrict__ qg, const int* __restrict__ kg,
                          const int* __restrict__ vg,
                          const float* __restrict__ qmin_g, const float* __restrict__ qscale_g,
                          const float* __restrict__ kmin_g, const float* __restrict__ kscale_g,
                          const float* __restrict__ vmin_g, const float* __restrict__ vscale_g,
                          float* __restrict__ out)
{
  __shared__ unsigned char Qu8[QT * DIM];
  __shared__ unsigned char Ku8[KT * DIM];
  __shared__ _Float16      Vt[DIM * KT];
  __shared__ _Float16      Pscr[NW * 16 * KT];
  __shared__ int           Qsp[2][QT], Ksp[2][KT];
  __shared__ float QminL[QT], QscaleL[QT];
  __shared__ float KminL[KT], KscaleL[KT], VminL[KT], VscaleL[KT];

  const int tid  = threadIdx.x;
  const int wave = tid >> 5;
  const int lid  = tid & 31;
  const int lg   = lid >> 4;
  const int ln   = lid & 15;
  const int bh   = blockIdx.y;
  const int q0   = blockIdx.x * QT;

  {
    const int r = tid >> 1, hf = tid & 1;
    const int s = q0 + r;
    const int4* p = (const int4*)(qg + ((size_t)s * BH + bh) * DIM + hf * 64);
    int acc = 0;
#pragma unroll
    for (int j = 0; j < 16; ++j) {
      int4 a = p[j];
      ((unsigned*)Qu8)[r * 32 + hf * 16 + j] = pack4(a);
      acc += a.x + a.y + a.z + a.w;
    }
    Qsp[hf][r] = acc;
    if (tid < QT) {
      const int ss = q0 + tid;
      QminL[tid]   = qmin_g[ss * BH + bh];
      QscaleL[tid] = qscale_g[ss * BH + bh];
    }
  }
  __syncthreads();

  float row_a[8], row_b[8], row_c[8];
#pragma unroll
  for (int r = 0; r < 8; ++r) {
    const int row = wave * 16 + r + lg * 8;
    const float qs = QscaleL[row];
    row_a[r] = qs;
    row_b[r] = qs * (float)(Qsp[0][row] + Qsp[1][row]);
    row_c[r] = QminL[row];
  }

  v8i qa[2];
#pragma unroll
  for (int kb = 0; kb < 2; ++kb) {
    const unsigned char* base = Qu8 + (wave * 16 + ln) * DIM + kb * 64;
#pragma unroll
    for (int v2 = 0; v2 < 4; ++v2) {
      const uint2 t = *(const uint2*)(base + (v2 >> 1) * 32 + (v2 & 1) * 8 + lg * 16);
      qa[kb][2 * v2] = (int)t.x; qa[kb][2 * v2 + 1] = (int)t.y;
    }
  }

  float m_[8], l_[8], pv_[8];
  v8f acc[8];
#pragma unroll
  for (int r = 0; r < 8; ++r) { m_[r] = -1e30f; l_[r] = 0.f; pv_[r] = 0.f; }
#pragma unroll
  for (int dt = 0; dt < 8; ++dt)
#pragma unroll
    for (int r = 0; r < 8; ++r) acc[dt][r] = 0.f;

  for (int kt = 0; kt < S_LEN / KT; ++kt) {
    const int kb0 = kt * KT;
    __syncthreads();
    {
      const int r = tid >> 1, hf = tid & 1;
      const int s = kb0 + r;
      const int4* kp = (const int4*)(kg + ((size_t)s * BH + bh) * DIM + hf * 64);
      int kacc = 0;
#pragma unroll
      for (int j = 0; j < 16; ++j) {
        int4 a = kp[j];
        ((unsigned*)Ku8)[r * 32 + hf * 16 + j] = pack4(a);
        kacc += a.x + a.y + a.z + a.w;
      }
      Ksp[hf][r] = kacc;

      const int kq = tid & 15;
#pragma unroll
      for (int i = 0; i < 4; ++i) {
        const int dq = (tid >> 4) * 4 + i;
        int4 a[4];
#pragma unroll
        for (int kk = 0; kk < 4; ++kk) {
          const int sk = kb0 + kq * 4 + kk;
          a[kk] = *(const int4*)(vg + ((size_t)sk * BH + bh) * DIM + dq * 4);
        }
#pragma unroll
        for (int dd = 0; dd < 4; ++dd) {
          const int d = dq * 4 + dd;
          h4 v;
          v.x = (_Float16)((&a[0].x)[dd]);
          v.y = (_Float16)((&a[1].x)[dd]);
          v.z = (_Float16)((&a[2].x)[dd]);
          v.w = (_Float16)((&a[3].x)[dd]);
          *(h4*)(Vt + d * KT + kq * 4) = v;
        }
      }

      if (tid < KT) {
        const int ss = kb0 + tid;
        KminL[tid]   = kmin_g[ss * BH + bh];
        KscaleL[tid] = kscale_g[ss * BH + bh];
        VminL[tid]   = vmin_g[ss * BH + bh];
        VscaleL[tid] = vscale_g[ss * BH + bh];
      }
      if (kt + 1 < S_LEN / KT) {
        __builtin_prefetch(kg + ((size_t)(kb0 + KT + r) * BH + bh) * DIM + hf * 64, 0, 0);
        __builtin_prefetch(vg + ((size_t)(kb0 + KT + r) * BH + bh) * DIM + hf * 64, 0, 0);
      }
    }
    __syncthreads();

    float sc[4][8];
#pragma unroll
    for (int nt = 0; nt < 4; ++nt) {
      v8i mm = {0, 0, 0, 0, 0, 0, 0, 0};
#pragma unroll
      for (int kb = 0; kb < 2; ++kb) {
        v8i kbf;
        const unsigned char* base = Ku8 + (nt * 16 + ln) * DIM + kb * 64;
#pragma unroll
        for (int v2 = 0; v2 < 4; ++v2) {
          const uint2 t = *(const uint2*)(base + (v2 >> 1) * 32 + (v2 & 1) * 8 + lg * 16);
          kbf[2 * v2] = (int)t.x; kbf[2 * v2 + 1] = (int)t.y;
        }
        mm = __builtin_amdgcn_wmma_i32_16x16x64_iu8(false, qa[kb], false, kbf, mm, false, false);
      }
      const int col = nt * 16 + ln;
      const float f_ks = KscaleL[col];
      const float f_km = KminL[col];
      const float f_kd = f_ks * (float)(Ksp[0][col] + Ksp[1][col]) + 128.0f * f_km;
#pragma unroll
      for (int r = 0; r < 8; ++r)
        sc[nt][r] = (row_a[r] * f_ks * (float)mm[r] + row_b[r] * f_km + row_c[r] * f_kd) * SCALE_LOG2;
    }

#pragma unroll
    for (int r = 0; r < 8; ++r) {
      float mx = fmaxf(fmaxf(sc[0][r], sc[1][r]), fmaxf(sc[2][r], sc[3][r]));
#pragma unroll
      for (int off = 8; off; off >>= 1) mx = fmaxf(mx, __shfl_xor(mx, off, 32));
      const float nm = fmaxf(m_[r], mx);
      const float cf = exp2f(m_[r] - nm);
      m_[r] = nm; l_[r] *= cf; pv_[r] *= cf;
#pragma unroll
      for (int dt = 0; dt < 8; ++dt) acc[dt][r] *= cf;
      float ps = 0.f, pvs = 0.f;
#pragma unroll
      for (int nt = 0; nt < 4; ++nt) {
        const int col = nt * 16 + ln;
        const float p = exp2f(sc[nt][r] - nm);
        ps  += p;
        pvs += p * VminL[col];
        Pscr[wave * (16 * KT) + (r + lg * 8) * KT + col] = (_Float16)(p * VscaleL[col]);
      }
#pragma unroll
      for (int off = 8; off; off >>= 1) {
        ps  += __shfl_xor(ps,  off, 32);
        pvs += __shfl_xor(pvs, off, 32);
      }
      l_[r] += ps; pv_[r] += pvs;
    }

    union PAU { v16h h; unsigned u[8]; };
    PAU pa[2];
#pragma unroll
    for (int kb = 0; kb < 2; ++kb) {
      const unsigned char* base = (const unsigned char*)Pscr + wave * (16 * KT * 2) + ln * (KT * 2) + kb * 64;
#pragma unroll
      for (int v2 = 0; v2 < 4; ++v2) {
        const uint2 t = *(const uint2*)(base + (v2 >> 1) * 32 + (v2 & 1) * 8 + lg * 16);
        pa[kb].u[2 * v2] = t.x; pa[kb].u[2 * v2 + 1] = t.y;
      }
    }
#pragma unroll
    for (int dt = 0; dt < 8; ++dt) {
#pragma unroll
      for (int kb = 0; kb < 2; ++kb) {
        PAU vb;
        const unsigned char* base = (const unsigned char*)Vt + (dt * 16 + ln) * (KT * 2) + kb * 64;
#pragma unroll
        for (int v2 = 0; v2 < 4; ++v2) {
          const uint2 t = *(const uint2*)(base + (v2 >> 1) * 32 + (v2 & 1) * 8 + lg * 16);
          vb.u[2 * v2] = t.x; vb.u[2 * v2 + 1] = t.y;
        }
        acc[dt] = __builtin_amdgcn_wmma_f32_16x16x32_f16(false, pa[kb].h, false, vb.h,
                                                         (short)0, acc[dt], false, false);
      }
    }
  }

#pragma unroll
  for (int r = 0; r < 8; ++r) {
    const float inv = 1.0f / l_[r];
    const int s = q0 + wave * 16 + r + lg * 8;
    float* op = out + ((size_t)s * BH + bh) * DIM;
#pragma unroll
    for (int dt = 0; dt < 8; ++dt)
      op[dt * 16 + ln] = (acc[dt][r] + pv_[r]) * inv;
  }
}

extern "C" void kernel_launch(void* const* d_in, const int* in_sizes, int n_in,
                              void* d_out, int out_size, void* d_ws, size_t ws_size,
                              hipStream_t stream) {
  (void)in_sizes; (void)n_in; (void)out_size; (void)d_ws; (void)ws_size;
  const int*   qg       = (const int*)d_in[0];
  const int*   kg       = (const int*)d_in[1];
  const int*   vg       = (const int*)d_in[2];
  const float* qmin_g   = (const float*)d_in[3];
  const float* qscale_g = (const float*)d_in[4];
  const float* kmin_g   = (const float*)d_in[5];
  const float* kscale_g = (const float*)d_in[6];
  const float* vmin_g   = (const float*)d_in[7];
  const float* vscale_g = (const float*)d_in[8];
  float* out = (float*)d_out;

  dim3 grid(S_LEN / QT, BH);
  dim3 block(128);
  flash_attn_u8_kernel<<<grid, block, 0, stream>>>(qg, kg, vg, qmin_g, qscale_g,
                                                   kmin_g, kscale_g, vmin_g, vscale_g, out);
}
